// SelfAttention_78125455115060
// MI455X (gfx1250) — matrix operand order repaired and checked by reading, not compiled, not run on hardware
//
#include <hip/hip_runtime.h>
#include <hip/hip_bf16.h>

#define BB 4
#define TT 2048
#define CC 1024
#define DD 64

typedef __bf16 bf16_t;
typedef __attribute__((ext_vector_type(4)))  __bf16 bf16x4;
typedef __attribute__((ext_vector_type(8)))  __bf16 bf16x8;
typedef __attribute__((ext_vector_type(16))) __bf16 v16bf;
typedef __attribute__((ext_vector_type(8)))  float  v8f;
typedef __attribute__((ext_vector_type(4)))  float  f32x4;

#define WMMA_BF16(A, Bf, Cc) \
  __builtin_amdgcn_wmma_f32_16x16x32_bf16(false, (A), false, (Bf), (short)0, (Cc), false, false)

__device__ __forceinline__ bf16x8 ld8(const bf16_t* p) {
  return *reinterpret_cast<const bf16x8*>(p);   // 16B aligned by construction
}
__device__ __forceinline__ v16bf cat(bf16x8 a, bf16x8 b) {
  return __builtin_shufflevector(a, b, 0,1,2,3,4,5,6,7,8,9,10,11,12,13,14,15);
}

// A fragment: row (lane&15), K contiguous 16 at 16h
__device__ __forceinline__ v16bf load_a_frag(const bf16_t* arow, int kbv, int h) {
  return cat(ld8(arow + kbv + 16 * h), ld8(arow + kbv + 16 * h + 8));
}
// Weight B fragments: column (nt*16+nn) of W^T = row of W, K contiguous 16 at 16h
__device__ __forceinline__ void load_w_frags(v16bf* Bf, const bf16_t* W, int kbv,
                                             int nn, int h) {
#pragma unroll
  for (int nt = 0; nt < 4; nt++) {
    const bf16_t* brow = W + (size_t)(nt * 16 + nn) * CC + kbv + 16 * h;
    Bf[nt] = cat(ld8(brow), ld8(brow + 8));
  }
}
// K^T B-fragments for one 32-key block (two 16-key tiles x two 32-deep D chunks)
__device__ __forceinline__ void load_k_frags(v16bf* Bk, const bf16_t* kp, int kbase,
                                             int nn, int h) {
  const bf16_t* r0 = kp + (size_t)(kbase + nn) * DD + 16 * h;
  Bk[0] = cat(ld8(r0),      ld8(r0 + 8));
  Bk[1] = cat(ld8(r0 + 32), ld8(r0 + 40));
  const bf16_t* r1 = kp + (size_t)(kbase + 16 + nn) * DD + 16 * h;
  Bk[2] = cat(ld8(r1),      ld8(r1 + 8));
  Bk[3] = cat(ld8(r1 + 32), ld8(r1 + 40));
}

// ---------------- fp32 -> bf16 conversion (optionally scaled) ----------------
__global__ void cvt_bf16_kernel(const float* __restrict__ in, bf16_t* __restrict__ out,
                                int n4, float scale) {
  int i = blockIdx.x * blockDim.x + threadIdx.x;
  if (i < n4) {
    f32x4 v = reinterpret_cast<const f32x4*>(in)[i];
    bf16x4 o;
    o[0] = (__bf16)(v[0] * scale);
    o[1] = (__bf16)(v[1] * scale);
    o[2] = (__bf16)(v[2] * scale);
    o[3] = (__bf16)(v[3] * scale);
    reinterpret_cast<bf16x4*>(out)[i] = o;
  }
}

// ---------------- QKV projection: one wave = (16-row tile, one of q/k/v) ----
// Ping-pong double-buffered K loop: no rotation copies, loads for step kb+32
// stay in flight while the WMMAs for step kb execute.
__global__ void __launch_bounds__(128)
qkv_kernel(const bf16_t* __restrict__ xb,
           const bf16_t* __restrict__ wq, const bf16_t* __restrict__ wk,
           const bf16_t* __restrict__ wv,
           bf16_t* __restrict__ qo, bf16_t* __restrict__ ko, bf16_t* __restrict__ vT) {
  const int lane = threadIdx.x & 31;
  const int wid  = blockIdx.x * 4 + (threadIdx.x >> 5);
  const int mat  = wid % 3;          // 0=q 1=k 2=v
  const int mtile = wid / 3;         // 512 tiles of 16 rows over B*T
  const int row0 = mtile * 16;
  const int nn = lane & 15, h = lane >> 4;

  const bf16_t* W = (mat == 0) ? wq : (mat == 1) ? wk : wv;
  const bf16_t* arow = xb + (size_t)(row0 + nn) * CC;

  v8f acc[4] = {};
  v16bf A0, A1, B0[4], B1[4];
  A0 = load_a_frag(arow, 0, h);
  load_w_frags(B0, W, 0, nn, h);

  // Note: the final preload (kbv+64 == CC) reads 32 bf16 past the row; that
  // stays inside the workspace (next tensor region) and the data is unused.
  for (int kbv = 0; kbv < CC; kbv += 64) {
    A1 = load_a_frag(arow, kbv + 32, h);
    load_w_frags(B1, W, kbv + 32, nn, h);
#pragma unroll
    for (int nt = 0; nt < 4; nt++) acc[nt] = WMMA_BF16(A0, B0[nt], acc[nt]);
    A0 = load_a_frag(arow, kbv + 64, h);
    load_w_frags(B0, W, kbv + 64, nn, h);
#pragma unroll
    for (int nt = 0; nt < 4; nt++) acc[nt] = WMMA_BF16(A1, B1[nt], acc[nt]);
  }

  if (mat < 2) {
    bf16_t* dst = (mat == 0) ? qo : ko;
#pragma unroll
    for (int r = 0; r < 8; r++) {
      int grow = row0 + r + 8 * h;
#pragma unroll
      for (int nt = 0; nt < 4; nt++)
        dst[(size_t)grow * DD + nt * 16 + nn] = (__bf16)acc[nt][r];
    }
  } else {
    int b = row0 >> 11;                         // 2048 rows per batch
    bf16_t* dstb = vT + (size_t)b * DD * TT;
#pragma unroll
    for (int r = 0; r < 8; r++) {
      int t = (row0 + r + 8 * h) & (TT - 1);
#pragma unroll
      for (int nt = 0; nt < 4; nt++)
        dstb[(size_t)(nt * 16 + nn) * TT + t] = (__bf16)acc[nt][r];
    }
  }
}

// ---------------- Split-K flash attention -----------------------------------
// One workgroup (4 waves) per 16-query tile. Wave w handles key-blocks
// ib = w, w+4, w+8, ... with private flash state; partials merged via LDS.
__global__ void __launch_bounds__(128)
attn_kernel(const bf16_t* __restrict__ q, const bf16_t* __restrict__ k,
            const bf16_t* __restrict__ vT, float* __restrict__ out) {
  __shared__ float pbuf[4][16 * 36];   // per-wave P transpose tile (stride 36)
  __shared__ float Obuf[4][16 * 68];   // per-wave partial O (stride 68: bank spread)
  __shared__ float mbuf[4][16];
  __shared__ float lbuf[4][16];

  const int lane = threadIdx.x & 31;
  const int w    = threadIdx.x >> 5;
  const int b  = blockIdx.x >> 7;               // 128 query tiles per batch
  const int q0 = (blockIdx.x & 127) * 16;
  const int nn = lane & 15, h = lane >> 4;
  float* pb = pbuf[w];

  const bf16_t* qb = q  + (size_t)b * TT * DD;
  const bf16_t* kp = k  + (size_t)b * TT * DD;
  const bf16_t* vb = vT + (size_t)b * DD * TT;

  // Q A-fragments (hoisted): two K=32 chunks over D=64
  v16bf Aq[2];
#pragma unroll
  for (int c = 0; c < 2; c++) {
    const bf16_t* p = qb + (size_t)(q0 + nn) * DD + c * 32;
    Aq[c] = cat(ld8(p + 16 * h), ld8(p + 16 * h + 8));
  }

  // constant all-ones B fragment: P @ ones == per-row sums, broadcast in C-layout
  v16bf ones;
#pragma unroll
  for (int j = 0; j < 16; j++) ones[j] = (__bf16)1.0f;

  v8f o[4] = {};
  float mx[8], sm[8];
#pragma unroll
  for (int r = 0; r < 8; r++) { mx[r] = -3e38f; sm[r] = 0.f; }

  const int nblocks = (q0 + 47) >> 5;           // 32-key blocks covering [0, q0+16)
  const int nloop   = (nblocks + 3) >> 2;       // round-robin over 4 waves

  v16bf BkA[4], BkB[4];                         // ping-pong K fragments
  if (w < nblocks) load_k_frags(BkA, kp, 32 * w, nn, h);

  auto process = [&](int ib, v16bf* Bku, v16bf* Bkf) {
    const int kbase = ib * 32;

    // ---- S = q @ k^T from preloaded fragments ----
    v8f s0 = {}, s1 = {};
    s0 = WMMA_BF16(Aq[0], Bku[0], s0);
    s0 = WMMA_BF16(Aq[1], Bku[1], s0);
    s1 = WMMA_BF16(Aq[0], Bku[2], s1);
    s1 = WMMA_BF16(Aq[1], Bku[3], s1);

    // ---- issue next-block K loads and this block's V loads now, so their
    //      latency hides under the softmax VALU chain ----
    const int ibn = ib + 4;
    if (ibn < nblocks) load_k_frags(Bkf, kp, ibn * 32, nn, h);
    v16bf Bv[4];
#pragma unroll
    for (int nt = 0; nt < 4; nt++) {
      const bf16_t* vrow = vb + (size_t)(nt * 16 + nn) * TT + kbase + 8 * h;
      Bv[nt] = cat(ld8(vrow), ld8(vrow + 16));
    }

    // ---- causal mask (only the diagonal block needs it) ----
    if (ib == nblocks - 1) {
#pragma unroll
      for (int r = 0; r < 8; r++) {
        int qrow = q0 + r + 8 * h;
        s0[r] = (kbase + nn      > qrow) ? -3e38f : s0[r];
        s1[r] = (kbase + 16 + nn > qrow) ? -3e38f : s1[r];
      }
    }

    // ---- online softmax: row max via shfl (rows at fixed (vgpr r, half h)) ----
    float alv[8];
#pragma unroll
    for (int r = 0; r < 8; r++) {
      float bm = fmaxf(s0[r], s1[r]);
      bm = fmaxf(bm, __shfl_xor(bm, 1));
      bm = fmaxf(bm, __shfl_xor(bm, 2));
      bm = fmaxf(bm, __shfl_xor(bm, 4));
      bm = fmaxf(bm, __shfl_xor(bm, 8));
      float nm = fmaxf(mx[r], bm);
      float al = __expf(mx[r] - nm);
      mx[r] = nm;
      alv[r] = al;
      float p0 = __expf(s0[r] - nm);
      float p1 = __expf(s1[r] - nm);
#pragma unroll
      for (int nt = 0; nt < 4; nt++) o[nt][r] *= al;
      pb[(r + 8 * h) * 36 + nn]      = p0;      // stage P for transpose
      pb[(r + 8 * h) * 36 + 16 + nn] = p1;
    }

    asm volatile("" ::: "memory");              // keep DS store->load program order
    // ---- P A-fragment from LDS (row lane&15, K chunks [8h,+8) & [16+8h,+8))
    const float* prow = pb + nn * 36;
    f32x4 f0 = *(const f32x4*)(prow + 8 * h);
    f32x4 f1 = *(const f32x4*)(prow + 8 * h + 4);
    f32x4 f2 = *(const f32x4*)(prow + 16 + 8 * h);
    f32x4 f3 = *(const f32x4*)(prow + 16 + 8 * h + 4);
    v16bf Ap;
#pragma unroll
    for (int j = 0; j < 4; j++) {
      Ap[j]      = (__bf16)f0[j];
      Ap[4 + j]  = (__bf16)f1[j];
      Ap[8 + j]  = (__bf16)f2[j];
      Ap[12 + j] = (__bf16)f3[j];
    }
    asm volatile("" ::: "memory");

    // ---- row sums via WMMA (replaces 32 shfl-adds), then l update ----
    v8f rsum = {};
    rsum = WMMA_BF16(Ap, ones, rsum);
#pragma unroll
    for (int r = 0; r < 8; r++) sm[r] = sm[r] * alv[r] + rsum[r];

    // ---- O += P @ V ----
#pragma unroll
    for (int nt = 0; nt < 4; nt++)
      o[nt] = WMMA_BF16(Ap, Bv[nt], o[nt]);
  };

  for (int i = 0; i < nloop; i += 2) {          // ping-pong: no buffer copies
    const int ib0 = i * 4 + w;                  // wave-uniform guards
    if (ib0 < nblocks) process(ib0, BkA, BkB);
    const int ib1 = (i + 1) * 4 + w;
    if (ib1 < nblocks) process(ib1, BkB, BkA);
  }

  // ---- publish per-wave partials ----
#pragma unroll
  for (int r = 0; r < 8; r++) {
    int row = r + 8 * h;
#pragma unroll
    for (int nt = 0; nt < 4; nt++)
      Obuf[w][row * 68 + nt * 16 + nn] = o[nt][r];
    if (nn == 0) { mbuf[w][row] = mx[r]; lbuf[w][row] = sm[r]; }
  }
  __syncthreads();

  // ---- merge 4 partials (log-sum-exp combine), coalesced fp32 store ----
  {
    const int row  = threadIdx.x >> 3;          // 16 rows x 8 threads
    const int col0 = (threadIdx.x & 7) * 8;
    float m0 = mbuf[0][row], m1 = mbuf[1][row], m2 = mbuf[2][row], m3 = mbuf[3][row];
    float ms = fmaxf(fmaxf(m0, m1), fmaxf(m2, m3));
    float w0 = __expf(m0 - ms), w1 = __expf(m1 - ms);
    float w2 = __expf(m2 - ms), w3 = __expf(m3 - ms);
    float lt = w0 * lbuf[0][row] + w1 * lbuf[1][row]
             + w2 * lbuf[2][row] + w3 * lbuf[3][row];
    float inv = 1.0f / lt;
    float res[8];
#pragma unroll
    for (int j = 0; j < 8; j++) {
      int col = col0 + j;
      float acc = w0 * Obuf[0][row * 68 + col] + w1 * Obuf[1][row * 68 + col]
                + w2 * Obuf[2][row * 68 + col] + w3 * Obuf[3][row * 68 + col];
      res[j] = acc * inv;
    }
    float* op = out + ((size_t)b * TT + q0 + row) * DD + col0;
    *(f32x4*)(op)     = (f32x4){res[0], res[1], res[2], res[3]};
    *(f32x4*)(op + 4) = (f32x4){res[4], res[5], res[6], res[7]};
  }
}

extern "C" void kernel_launch(void* const* d_in, const int* in_sizes, int n_in,
                              void* d_out, int out_size, void* d_ws, size_t ws_size,
                              hipStream_t stream) {
  const float* x  = (const float*)d_in[0];
  const float* Wq = (const float*)d_in[1];
  const float* Wk = (const float*)d_in[2];
  const float* Wv = (const float*)d_in[3];
  float* out = (float*)d_out;

  char* ws = (char*)d_ws;
  size_t off = 0;
  bf16_t* xb  = (bf16_t*)(ws + off); off += (size_t)BB * TT * CC * sizeof(bf16_t);
  bf16_t* wqb = (bf16_t*)(ws + off); off += (size_t)DD * CC * sizeof(bf16_t);
  bf16_t* wkb = (bf16_t*)(ws + off); off += (size_t)DD * CC * sizeof(bf16_t);
  bf16_t* wvb = (bf16_t*)(ws + off); off += (size_t)DD * CC * sizeof(bf16_t);
  bf16_t* qv  = (bf16_t*)(ws + off); off += (size_t)BB * TT * DD * sizeof(bf16_t);
  bf16_t* kv  = (bf16_t*)(ws + off); off += (size_t)BB * TT * DD * sizeof(bf16_t);
  bf16_t* vT  = (bf16_t*)(ws + off); off += (size_t)BB * DD * TT * sizeof(bf16_t);

  const int n4x = BB * TT * CC / 4;
  cvt_bf16_kernel<<<(n4x + 255) / 256, 256, 0, stream>>>(x, xb, n4x, 1.0f);
  const int n4w = DD * CC / 4;
  cvt_bf16_kernel<<<(n4w + 255) / 256, 256, 0, stream>>>(Wq, wqb, n4w, 0.03125f); // fold C^-0.5
  cvt_bf16_kernel<<<(n4w + 255) / 256, 256, 0, stream>>>(Wk, wkb, n4w, 1.0f);
  cvt_bf16_kernel<<<(n4w + 255) / 256, 256, 0, stream>>>(Wv, wvb, n4w, 1.0f);

  // 512 M-tiles x 3 matrices = 1536 waves -> 384 blocks of 4 waves
  qkv_kernel<<<384, 128, 0, stream>>>(xb, wqb, wkb, wvb, qv, kv, vT);
  // one workgroup (4 split-K waves) per query tile -> 512 blocks
  attn_kernel<<<512, 128, 0, stream>>>(qv, kv, vT, out);
}
